// AxialAttentionNaiveAISummer_26147760898128
// MI455X (gfx1250) — matrix operand order repaired and checked by reading, not compiled, not run on hardware
//
#include <hip/hip_runtime.h>

// Fused axial attention for MI455X (gfx1250, wave32, WMMA + TDM).
// One workgroup (256 threads = 8 waves) per batch element; all intermediates
// in 264KB dynamic LDS (< 320KB/WGP); every contraction runs on
// v_wmma_f32_16x16x32_f16 with f32 accumulation; x_in tile staged via the
// Tensor Data Mover (tensor_load_to_lds + s_wait_tensorcnt).

#define HEADS 8
#define DIMX  128
#define INCH  128

typedef _Float16 h16;
typedef __attribute__((ext_vector_type(16))) _Float16 v16h;
typedef __attribute__((ext_vector_type(8)))  _Float16 v8h;
typedef __attribute__((ext_vector_type(8)))  float    v8f;
typedef __attribute__((ext_vector_type(4)))  unsigned int u32x4;
typedef __attribute__((ext_vector_type(8)))  int          i32x8;
typedef __attribute__((ext_vector_type(4)))  int          i32x4;

// ---- workspace layout (bytes) ----
// [0, 65536)      : conv_w pre-swizzled to f16 WMMA A-fragment order
// [65536, 66560)  : BN inv   (256 f32)
// [66560, 67584)  : BN bias  (256 f32)

// ---- LDS byte offsets ----
#define L_QKT   0        // [h][d][16] h16 : q(i<8), k(i-8) transposed   (32768)
#define L_V     32768    // [h][i][136] h16: v row-major over j          (34816)
#define L_RELRM 67584    // [i<32][256] h16: relative row-major          (16384)
#define L_RELT  83968    // [t<256][32] h16: relative transposed         (16384)
#define L_ATTN  100352   // [d][136] h16   : softmax rows                (34816)
#define L_LOGIT 135168   // [d][132] f32   : logits (phase1: sXT [d][136])(67584)
#define L_SKEW  202752   // [d][264] h16   : skewed attn (phase0: raw x) (67584)
#define L_TOTAL 270336

__device__ __forceinline__ v8f wmma16(v16h a, v16h b, v8f c) {
  return __builtin_amdgcn_wmma_f32_16x16x32_f16(false, a, false, b, (short)0, c,
                                                false, false);
}
__device__ __forceinline__ v8f zero8f() {
  v8f z;
#pragma unroll
  for (int i = 0; i < 8; ++i) z[i] = 0.f;
  return z;
}
// two 16B chunks (ds_read_b128 / global_load_b128) -> one 16-half fragment
__device__ __forceinline__ v16h ld2x8(const h16* p0, const h16* p1) {
  v8h a = *(const v8h*)p0;
  v8h b = *(const v8h*)p1;
  v16h r;
#pragma unroll
  for (int i = 0; i < 8; ++i) { r[i] = a[i]; r[i + 8] = b[i]; }
  return r;
}
// fragment with low 8 halves from LDS (lane-masked select), high 8 zero
__device__ __forceinline__ v16h ldlo8_sel(const h16* p, bool keep) {
  v8h a = *(const v8h*)p;
  v16h r;
#pragma unroll
  for (int i = 0; i < 8; ++i) {
    r[i]     = keep ? a[i] : (h16)0.f;
    r[i + 8] = (h16)0.f;
  }
  return r;
}

// ---------------- prep kernels ----------------

__global__ void prep_bn(const float* __restrict__ g, const float* __restrict__ be,
                        const float* __restrict__ mu, const float* __restrict__ var,
                        float* __restrict__ inv, float* __restrict__ bias) {
  int o = threadIdx.x;  // 0..255
  float iv = g[o] * rsqrtf(var[o] + 1e-5f);
  float bs = be[o] - mu[o] * iv;
  if (o < 64) {                      // q channels: fold DIM^-0.5 scale
    const float s = 0.08838834764831845f;
    iv *= s; bs *= s;
  }
  inv[o] = iv; bias[o] = bs;
}

// conv_w (256x128 f32) -> f16 in WMMA A-operand order:
// [mtile][kstep][lane][16 halves]; lane<16: M=mt*16+lane, halves K{0..7,16..23};
// lane>=16: M=mt*16+lane-16, halves K{8..15,24..31} (K base = ks*32).
__global__ void prep_w(const float* __restrict__ w, h16* __restrict__ wA) {
  int idx  = blockIdx.x * 256 + threadIdx.x;   // exactly 32768 total
  int e    = idx & 15;
  int lane = (idx >> 4) & 31;
  int ks   = (idx >> 9) & 3;
  int mt   = idx >> 11;
  int M = mt * 16 + (lane & 15);
  int K = ks * 32 + ((lane >> 4) << 3) + ((e >> 3) << 4) + (e & 7);
  wA[idx] = (h16)w[M * 128 + K];
}

// ---------------- fused main kernel ----------------

__global__ void __launch_bounds__(256)
axial_kernel(const float* __restrict__ x_in, const float* __restrict__ relative,
             const h16* __restrict__ wA, const float* __restrict__ inv,
             const float* __restrict__ bias, float* __restrict__ out) {
  extern __shared__ char smem[];
  h16*   sQKT   = (h16*)(smem + L_QKT);
  h16*   sV     = (h16*)(smem + L_V);
  h16*   sRelRM = (h16*)(smem + L_RELRM);
  h16*   sRelT  = (h16*)(smem + L_RELT);
  h16*   sAttn  = (h16*)(smem + L_ATTN);
  float* sLogit = (float*)(smem + L_LOGIT);
  h16*   sXT    = (h16*)(smem + L_LOGIT);   // phase-1 only overlay
  h16*   sSkew  = (h16*)(smem + L_SKEW);

  const int tid  = threadIdx.x;
  const int lane = tid & 31;
  // wave id as a compiler-provable scalar -> wave-level branches stay on SALU
  const int wvu  = __builtin_amdgcn_readfirstlane(tid >> 5);
  const int lr   = lane & 15;       // row/col within tile
  const int lh   = lane >> 4;       // lane-half select
  const int b    = blockIdx.x;
  const bool lo  = (lane < 16);

  const float* xb = x_in + (size_t)b * (INCH * DIMX);

  __builtin_prefetch(wA + (size_t)lane * 1024, 0, 1);   // warm swizzled weights

#if defined(__gfx1250__) && __has_builtin(__builtin_amdgcn_tensor_load_to_lds)
  // ---- TDM: DMA the 128x128 f32 x_in tile into LDS (region later = sSkew) ----
  if (wvu == 0) {
    unsigned long long ga = (unsigned long long)xb;
    u32x4 g0 = { 1u,                                   // count=1, user desc
                 (unsigned int)L_SKEW,                 // lds_addr (bytes)
                 (unsigned int)ga,                     // global_addr[31:0]
                 (unsigned int)((ga >> 32) & 0x01FFFFFFull) | (2u << 30) }; // [56:32]|type=2
    i32x8 g1 = { (int)(2u << 16),        // data_size=4B, wg_mask=0
                 (int)(128u << 16),      // tensor_dim0[15:0]=128 (bits 63:48)
                 (int)(128u << 16),      // tensor_dim1[15:0]=128 (bits 95:80)
                 (int)(128u << 16),      // tile_dim0=128 (bits 127:112)
                 128,                    // tile_dim1=128, tile_dim2=0
                 128,                    // tensor_dim0_stride=128
                 (int)(16384u << 16),    // tensor_dim1_stride[15:0]
                 0 };
    i32x4 gz = { 0, 0, 0, 0 };
#if __clang_major__ >= 23
    i32x8 gz8 = { 0, 0, 0, 0, 0, 0, 0, 0 };
    __builtin_amdgcn_tensor_load_to_lds(g0, g1, gz, gz, gz8, 0);
#else
    __builtin_amdgcn_tensor_load_to_lds(g0, g1, gz, gz, 0);
#endif
  }
#endif

  // stage relative (32x255 f32) into LDS in both orientations, pad t=255 with 0
  for (int idx = tid; idx < 32 * 256; idx += 256) {
    int i = idx >> 8, t = idx & 255;
    h16 v = (h16)0.f;
    if (t < 255) v = (h16)relative[i * 255 + t];
    sRelRM[i * 256 + t] = v;
    sRelT[t * 32 + i]   = v;
  }

#if defined(__gfx1250__) && __has_builtin(__builtin_amdgcn_tensor_load_to_lds)
  if (wvu == 0) __builtin_amdgcn_s_wait_tensorcnt(0);
  __syncthreads();
  {
    const float* ldsX = (const float*)(smem + L_SKEW);
    for (int idx = tid; idx < INCH * DIMX; idx += 256) {
      int c = idx >> 7, d = idx & 127;
      sXT[d * 136 + c] = (h16)ldsX[idx];   // f32 -> f16, transpose [d][c]
    }
  }
#else
  for (int idx = tid; idx < INCH * DIMX; idx += 256) {
    int c = idx >> 7, d = idx & 127;
    sXT[d * 136 + c] = (h16)xb[idx];
  }
#endif
  __syncthreads();

  // zero the skew buffer ONCE: every head rewrites the same band t in [d,d+127],
  // so out-of-band zeros survive across the head loop. (Raw x overlay is dead now.)
  for (int idx = tid; idx < (128 * 264) / 2; idx += 256)
    ((unsigned int*)sSkew)[idx] = 0u;

  // ---- Phase 1: Y = W @ X  (M=256,N=128,K=128) + BN; scatter q/k^T and v ----
  // For fragment row r: head hh = r, qkv-channel c = 2*mt + lh (mt is scalar).
  for (int mi = 0; mi < 2; ++mi) {
    const int mt = wvu * 2 + mi;
    v16h aw[4];
#pragma unroll
    for (int ks = 0; ks < 4; ++ks) {
      const h16* p = wA + (size_t)((mt * 4 + ks) * 32 + lane) * 16;
      aw[ks] = ld2x8(p, p + 8);                 // pre-swizzled: contiguous 32B
    }
    float invr[8], biasr[8];
#pragma unroll
    for (int r = 0; r < 8; ++r) {
      int o = mt * 16 + r + (lh << 3);
      invr[r] = inv[o]; biasr[r] = bias[o];
    }
    const int c2 = 2 * mt + lh;                 // = o>>3 for every r
#pragma unroll
    for (int nt = 0; nt < 8; ++nt) {
      v8f acc = zero8f();
#pragma unroll
      for (int ks = 0; ks < 4; ++ks) {
        const h16* bp = sXT + (nt * 16 + lr) * 136 + ks * 32 + lh * 8;
        acc = wmma16(aw[ks], ld2x8(bp, bp + 16), acc);
      }
      const int d = nt * 16 + lr;
      if (mt < 8) {          // scalar branch: q/k rows -> transposed store
#pragma unroll
        for (int r = 0; r < 8; ++r)
          sQKT[(r * 128 + d) * 16 + c2] = (h16)(acc[r] * invr[r] + biasr[r]);
      } else {               // v rows -> row-major store
#pragma unroll
        for (int r = 0; r < 8; ++r)
          sV[(r * 16 + (c2 - 16)) * 136 + d] = (h16)(acc[r] * invr[r] + biasr[r]);
      }
    }
  }
  __syncthreads();

  // ---- Phase 2: per-head attention ----
  for (int h = 0; h < HEADS; ++h) {
    const int dt = wvu;  // this wave owns d-tile dt

    // dots[d,j] = sum_i q[i,d] k[i,j]  (K=8 padded to 32)
    {
      v16h aq = ldlo8_sel(sQKT + (h * 128 + dt * 16 + lr) * 16, lo);
#pragma unroll
      for (int jt = 0; jt < 8; ++jt) {
        v16h bk = ldlo8_sel(sQKT + (h * 128 + jt * 16 + lr) * 16 + 8, lo);
        v8f acc = wmma16(aq, bk, zero8f());
#pragma unroll
        for (int r = 0; r < 8; ++r)
          sLogit[(dt * 16 + r + (lh << 3)) * 132 + jt * 16 + lr] = acc[r];
      }
    }
    __syncthreads();

    // R[d,t] = sum_{i<16} [q;k][i,d]*rel_qk[i,t]; scatter-add at j=d+127-t.
    // For tile tti, j in [112-16*tti, 142-16*tti]: only tti==0 / tti==8 can
    // fall outside [0,127] -> bounds checks only on the two edge tiles.
    {
      v16h ar = ldlo8_sel(sQKT + (h * 128 + dt * 16 + lr) * 16 + lh * 8, true);
#pragma unroll
      for (int tti = 0; tti < 9; ++tti) {
        int t0 = (dt + tti) * 16;
        const h16* p = sRelT + (t0 + lr) * 32;
        v8h x0 = *(const v8h*)p, x1 = *(const v8h*)(p + 8);
        v16h br;
#pragma unroll
        for (int i = 0; i < 8; ++i) {
          br[i]     = lo ? x0[i] : x1[i];
          br[i + 8] = (h16)0.f;
        }
        v8f acc = wmma16(ar, br, zero8f());
#pragma unroll
        for (int r = 0; r < 8; ++r) {
          int d = dt * 16 + r + (lh << 3);
          int j = d + 127 - (t0 + lr);
          if (tti == 0)      { if (j < 128) sLogit[d * 132 + j] += acc[r]; }
          else if (tti == 8) { if (j >= 0)  sLogit[d * 132 + j] += acc[r]; }
          else               {              sLogit[d * 132 + j] += acc[r]; }
        }
      }
    }
    __syncthreads();

    // softmax rows; emit attn f16 row-major AND its skewed copy for the kv GEMM
    if (tid < 128) {
      float* L = sLogit + tid * 132;
      float mx = -3.4e38f;
      for (int j = 0; j < 128; ++j) mx = fmaxf(mx, L[j]);
      float s = 0.f;
      for (int j = 0; j < 128; ++j) { float e = __expf(L[j] - mx); s += e; L[j] = e; }
      float rs = 1.f / s;
      for (int j = 0; j < 128; ++j) {
        h16 v = (h16)(L[j] * rs);
        sAttn[tid * 136 + j] = v;
        sSkew[tid * 264 + tid + 127 - j] = v;
      }
    }
    __syncthreads();

    // out[i,d] = v @ attn^T  and  kv[i,d] = rel_v @ skew^T, fused accumulator
    {
      const int nt = wvu;  // this wave owns d-tile nt of the output
      v8f acc = zero8f();
#pragma unroll
      for (int ks = 0; ks < 4; ++ks) {  // K = j (128)
        const h16* ap = sV + (h * 16 + lr) * 136 + ks * 32 + lh * 8;
        const h16* bp = sAttn + (nt * 16 + lr) * 136 + ks * 32 + lh * 8;
        acc = wmma16(ld2x8(ap, ap + 16), ld2x8(bp, bp + 16), acc);
      }
#pragma unroll
      for (int ks = 0; ks < 8; ++ks) {  // K = t (256)
        const h16* ap = sRelRM + (16 + lr) * 256 + ks * 32 + lh * 8;
        const h16* bp = sSkew + (nt * 16 + lr) * 264 + ks * 32 + lh * 8;
        acc = wmma16(ld2x8(ap, ap + 16), ld2x8(bp, bp + 16), acc);
      }
      float* ob = out + (size_t)b * (128 * 128) + (size_t)(h * 16) * 128 + nt * 16 + lr;
#pragma unroll
      for (int r = 0; r < 8; ++r) ob[(size_t)(r + (lh << 3)) * 128] = acc[r];
    }
    __syncthreads();
  }
}

extern "C" void kernel_launch(void* const* d_in, const int* in_sizes, int n_in,
                              void* d_out, int out_size, void* d_ws, size_t ws_size,
                              hipStream_t stream) {
  (void)n_in; (void)out_size; (void)ws_size;
  const float* x_in     = (const float*)d_in[0];
  const float* conv_w   = (const float*)d_in[1];
  const float* bn_gamma = (const float*)d_in[2];
  const float* bn_beta  = (const float*)d_in[3];
  const float* bn_mean  = (const float*)d_in[4];
  const float* bn_var   = (const float*)d_in[5];
  const float* relative = (const float*)d_in[6];
  float* out = (float*)d_out;

  char* ws = (char*)d_ws;
  h16*   wA   = (h16*)ws;               // 64 KB swizzled f16 weights
  float* inv  = (float*)(ws + 65536);   // 1 KB
  float* bias = (float*)(ws + 66560);   // 1 KB

  const int B = in_sizes[0] / (INCH * DIMX);  // 2048

  prep_bn<<<1, 256, 0, stream>>>(bn_gamma, bn_beta, bn_mean, bn_var, inv, bias);
  prep_w<<<128, 256, 0, stream>>>(conv_w, wA);

  // 264 KB dynamic LDS (< 320 KB per WGP on CDNA5).
  axial_kernel<<<B, 256, (size_t)L_TOTAL, stream>>>(x_in, relative, wA, inv,
                                                    bias, out);
}
